// GCN_56478819943012
// MI455X (gfx1250) — compile-verified, matrix operand order checked by reading, not run on hardware
//
#include <hip/hip_runtime.h>
#include <hip/hip_bf16.h>

// ---------------------------------------------------------------------------
// GCN (3x GCNConv + MLP head + global mean pool) for MI455X / gfx1250.
//   - Dense transforms via V_WMMA_F32_16X16X32_BF16 (wave32, fp32 accum)
//   - Weights pre-swizzled once into per-lane WMMA B-fragment layout (bf16)
//     -> hot loop B load = one contiguous 32B vector load per lane
//   - A fragments: 4x b128 loads + v_perm_b32 bf16 packing
//   - Edge aggregation via coalesced float4 gather + global_atomic_add_f32
// ---------------------------------------------------------------------------

typedef __attribute__((ext_vector_type(16))) __bf16    v16bf;
typedef __attribute__((ext_vector_type(8)))  float     v8f;
typedef __attribute__((ext_vector_type(8)))  unsigned  v8u;

#define N_GRAPHS 256

__device__ __forceinline__ __bf16 f2bf(float f) {
    // round-to-nearest-even float32 -> bf16 (one-time weight pack)
    union { float f; unsigned u; } v; v.f = f;
    unsigned r = v.u + 0x7FFFu + ((v.u >> 16) & 1u);
    unsigned short h = (unsigned short)(r >> 16);
    __bf16 b;
    __builtin_memcpy(&b, &h, 2);
    return b;
}

// two floats -> packed bf16x2 dword: +0x8000 (round-to-nearest) then v_perm_b32
__device__ __forceinline__ unsigned pk2bf(float lo, float hi) {
    unsigned a = __float_as_uint(lo) + 0x8000u;
    unsigned b = __float_as_uint(hi) + 0x8000u;
    return __builtin_amdgcn_perm(b, a, 0x07060302u);  // {b[31:16], a[31:16]}
}

// --------------------------- degree / normalization ------------------------

__global__ void gcn_deg_init(float* __restrict__ deg, int n) {
    int i = blockIdx.x * blockDim.x + threadIdx.x;
    if (i < n) deg[i] = 1.0f;                      // self-loop contributes 1
}

__global__ void gcn_deg_count(float* __restrict__ deg, const int* __restrict__ dst, int e) {
    int i = blockIdx.x * blockDim.x + threadIdx.x;
    if (i < e) atomicAdd(&deg[dst[i]], 1.0f);
}

__global__ void gcn_inv_sqrt(float* __restrict__ deg, int n) {
    int i = blockIdx.x * blockDim.x + threadIdx.x;
    if (i < n) deg[i] = __frsqrt_rn(deg[i]);       // deg >= 1 always
}

// -------------------------- weight pre-swizzle -----------------------------
// Wp layout: [kTile][nTile][lane 0..31][16 bf16]; lane half = K+16 sub-block,
// lane&15 = column. Per-lane WMMA B fragment becomes one 32B contiguous load.

__global__ void gcn_pack_w(const float* __restrict__ W, __bf16* __restrict__ Wp,
                           int K, int Nout) {
    int idx = blockIdx.x * blockDim.x + threadIdx.x;   // (kt*nTiles+nt)*32+lane
    int total = (K >> 5) * (Nout >> 4) * 32;
    if (idx >= total) return;
    int lane = idx & 31, tile = idx >> 5;
    int nTiles = Nout >> 4;
    int kt = tile / nTiles, nt = tile - kt * nTiles;
    int half = lane >> 4, l16 = lane & 15;
    const float* src = W + (size_t)(kt * 32 + half * 16) * Nout + nt * 16 + l16;
    __bf16* dst = Wp + (size_t)idx * 16;               // contiguous across grid
#pragma unroll
    for (int e = 0; e < 16; ++e) dst[e] = f2bf(src[(size_t)e * Nout]);
}

// ------------------------------- WMMA GEMM ---------------------------------
// C[M,Nout] = act( A[M,K] @ W[K,Nout] + bias ). Wave tile = 32(M) x 64(N),
// block = (32,4) -> 128 rows/block. grid = (ceil(M/128), Nout/64).

template<int RELU>
__device__ __forceinline__ v16bf load_a_frag(const float* __restrict__ p) {
    const float4* q = (const float4*)p;
    v8u pk;
#pragma unroll
    for (int j = 0; j < 4; ++j) {
        float4 f = q[j];
        if (RELU) {
            f.x = fmaxf(f.x, 0.f); f.y = fmaxf(f.y, 0.f);
            f.z = fmaxf(f.z, 0.f); f.w = fmaxf(f.w, 0.f);
        }
        pk[2 * j]     = pk2bf(f.x, f.y);
        pk[2 * j + 1] = pk2bf(f.z, f.w);
    }
    return __builtin_bit_cast(v16bf, pk);
}

template<int RELU_IN, int RELU_OUT>
__global__ void __launch_bounds__(128)
gcn_gemm_wmma(const float* __restrict__ A, const __bf16* __restrict__ Wp,
              const float* __restrict__ bias, float* __restrict__ C,
              int M, int K, int Nout) {
    const int lane  = threadIdx.x;        // wave32
    const int half  = lane >> 4;
    const int l16   = lane & 15;
    const int mBase = (blockIdx.x * 4 + threadIdx.y) * 32;
    const int nTiles = Nout >> 4;
    const int nt0   = blockIdx.y * 4;
    const int nBase = nt0 * 16;

    v8f acc[2][4] = {};

    int r0 = mBase + l16;      if (r0 > M - 1) r0 = M - 1;  // clamp; never stored
    int r1 = mBase + 16 + l16; if (r1 > M - 1) r1 = M - 1;
    const float* __restrict__ a0p = A + (size_t)r0 * K + half * 16;
    const float* __restrict__ a1p = A + (size_t)r1 * K + half * 16;
    const int kTiles = K >> 5;

    for (int kt = 0; kt < kTiles; ++kt) {
        v16bf a0 = load_a_frag<RELU_IN>(a0p + kt * 32);
        v16bf a1 = load_a_frag<RELU_IN>(a1p + kt * 32);
        const __bf16* __restrict__ bp =
            Wp + ((size_t)(kt * nTiles + nt0) * 32 + lane) * 16;
#pragma unroll
        for (int t = 0; t < 4; ++t) {
            v16bf b = *(const v16bf*)(bp + (size_t)t * 32 * 16);  // 32B/lane, coalesced
            acc[0][t] = __builtin_amdgcn_wmma_f32_16x16x32_bf16(
                false, a0, false, b, (short)0, acc[0][t], false, false);
            acc[1][t] = __builtin_amdgcn_wmma_f32_16x16x32_bf16(
                false, a1, false, b, (short)0, acc[1][t], false, false);
        }
    }

#pragma unroll
    for (int s = 0; s < 2; ++s) {
#pragma unroll
        for (int r = 0; r < 8; ++r) {
            const int row = mBase + s * 16 + half * 8 + r;
            if (row < M) {
#pragma unroll
                for (int t = 0; t < 4; ++t) {
                    const int col = nBase + t * 16 + l16;
                    float v = acc[s][t][r];
                    if (bias) v += bias[col];
                    if (RELU_OUT) v = fmaxf(v, 0.0f);
                    C[(size_t)row * Nout + col] = v;
                }
            }
        }
    }
}

// --------------------------- aggregation (A_hat @ t) -----------------------

// h[i,f] = b[f] + t[i,f] * inv[i]^2     (self-loop term; fully overwrites h)
__global__ void gcn_agg_init(const float* __restrict__ t, const float* __restrict__ bias,
                             const float* __restrict__ inv, float* __restrict__ h,
                             int n, int H) {
    const int Hc = H >> 2;
    long long idx = (long long)blockIdx.x * blockDim.x + threadIdx.x;
    if (idx >= (long long)n * Hc) return;
    int i = (int)(idx / Hc), f4 = (int)(idx % Hc);
    float w = inv[i]; w *= w;
    float4 tv = *(const float4*)(t + idx * 4);
    float4 bv = *(const float4*)(bias + f4 * 4);
    float4 o;
    o.x = bv.x + tv.x * w; o.y = bv.y + tv.y * w;
    o.z = bv.z + tv.z * w; o.w = bv.w + tv.w * w;
    *(float4*)(h + idx * 4) = o;
}

// h[dst] += t[src] * inv[src]*inv[dst]  (float4 per thread, consecutive lanes
// cover consecutive features of one edge -> coalesced gather & scatter)
__global__ void gcn_edge_scatter(const float* __restrict__ t, float* __restrict__ h,
                                 const int* __restrict__ src, const int* __restrict__ dst,
                                 const float* __restrict__ inv, int e, int H) {
    const int Hc = H >> 2;
    long long idx = (long long)blockIdx.x * blockDim.x + threadIdx.x;
    if (idx >= (long long)e * Hc) return;
    int ed = (int)(idx / Hc), f4 = (int)(idx % Hc);
    int s = src[ed], d = dst[ed];
    float w = inv[s] * inv[d];
    const float4 v = *(const float4*)(t + (size_t)s * H + f4 * 4);
    float* out = h + (size_t)d * H + f4 * 4;
    atomicAdd(out + 0, v.x * w);
    atomicAdd(out + 1, v.y * w);
    atomicAdd(out + 2, v.z * w);
    atomicAdd(out + 3, v.w * w);
}

// ------------------------- head (512->4) + mean pool -----------------------

__global__ void gcn_zero(float* __restrict__ p, int n) {
    int i = blockIdx.x * blockDim.x + threadIdx.x;
    if (i < n) p[i] = 0.0f;
}

// one wave32 per node: lanes stride K=512, shfl-reduce, lane0 scatters to graph
__global__ void gcn_head_pool(const float* __restrict__ z, const float* __restrict__ lw3,
                              const float* __restrict__ lb3, const int* __restrict__ batch,
                              float* __restrict__ sums, float* __restrict__ counts, int n) {
    int node = (blockIdx.x * blockDim.x + threadIdx.x) >> 5;
    int lane = threadIdx.x & 31;
    if (node >= n) return;
    const float* __restrict__ row = z + (size_t)node * 512;
    float a0 = 0.f, a1 = 0.f, a2 = 0.f, a3 = 0.f;
    for (int k = lane; k < 512; k += 32) {
        float v = row[k];
        const float4 w = *(const float4*)(lw3 + k * 4);
        a0 += v * w.x; a1 += v * w.y; a2 += v * w.z; a3 += v * w.w;
    }
#pragma unroll
    for (int off = 16; off > 0; off >>= 1) {
        a0 += __shfl_down(a0, off, 32);
        a1 += __shfl_down(a1, off, 32);
        a2 += __shfl_down(a2, off, 32);
        a3 += __shfl_down(a3, off, 32);
    }
    if (lane == 0) {
        int g = batch[node];
        atomicAdd(&sums[g * 4 + 0], a0 + lb3[0]);
        atomicAdd(&sums[g * 4 + 1], a1 + lb3[1]);
        atomicAdd(&sums[g * 4 + 2], a2 + lb3[2]);
        atomicAdd(&sums[g * 4 + 3], a3 + lb3[3]);
        atomicAdd(&counts[g], 1.0f);
    }
}

__global__ void gcn_pool_div(const float* __restrict__ sums, const float* __restrict__ counts,
                             float* __restrict__ out) {
    int i = blockIdx.x * blockDim.x + threadIdx.x;
    if (i < N_GRAPHS * 4) out[i] = sums[i] / fmaxf(counts[i >> 2], 1.0f);
}

// ------------------------------- launcher ----------------------------------

static inline unsigned cdiv_ll(long long a, long long b) { return (unsigned)((a + b - 1) / b); }

extern "C" void kernel_launch(void* const* d_in, const int* in_sizes, int n_in,
                              void* d_out, int out_size, void* d_ws, size_t ws_size,
                              hipStream_t stream) {
    const float* x    = (const float*)d_in[0];
    const int*   ei   = (const int*)d_in[1];
    const int*   bat  = (const int*)d_in[2];
    const float* W1   = (const float*)d_in[3];   const float* b1  = (const float*)d_in[4];
    const float* W2   = (const float*)d_in[5];   const float* b2  = (const float*)d_in[6];
    const float* W3   = (const float*)d_in[7];   const float* b3  = (const float*)d_in[8];
    const float* lw1  = (const float*)d_in[9];   const float* lb1 = (const float*)d_in[10];
    const float* lw2  = (const float*)d_in[11];  const float* lb2 = (const float*)d_in[12];
    const float* lw3  = (const float*)d_in[13];  const float* lb3 = (const float*)d_in[14];

    const int N = in_sizes[0] / 128;      // nodes
    const int E = in_sizes[1] / 2;        // edges
    const int* src = ei;                  // edge_index[0]
    const int* dst = ei + E;              // edge_index[1]

    // workspace: inv[N] | P[N*1024] | Q[N*512] | sums | counts | packed weights
    float* inv    = (float*)d_ws;
    float* P      = inv + (size_t)N;
    float* Q      = P + (size_t)N * 1024;
    float* sums   = Q + (size_t)N * 512;
    float* counts = sums + N_GRAPHS * 4;
    __bf16* W1p   = (__bf16*)(counts + N_GRAPHS);
    __bf16* W2p   = W1p + 128 * 128;
    __bf16* W3p   = W2p + 128 * 256;
    __bf16* L1p   = W3p + 256 * 512;
    __bf16* L2p   = L1p + 512 * 1024;

    const dim3 gblk(32, 4, 1);

    // 0. pre-swizzle weights into WMMA B-fragment layout (bf16, ~2.5MB total)
    gcn_pack_w<<<cdiv_ll(128 * 128 / 16, 256), 256, 0, stream>>>(W1,  W1p, 128, 128);
    gcn_pack_w<<<cdiv_ll(128 * 256 / 16, 256), 256, 0, stream>>>(W2,  W2p, 128, 256);
    gcn_pack_w<<<cdiv_ll(256 * 512 / 16, 256), 256, 0, stream>>>(W3,  W3p, 256, 512);
    gcn_pack_w<<<cdiv_ll(512 * 1024 / 16, 256), 256, 0, stream>>>(lw1, L1p, 512, 1024);
    gcn_pack_w<<<cdiv_ll(1024 * 512 / 16, 256), 256, 0, stream>>>(lw2, L2p, 1024, 512);

    // 1. normalization: deg (self-loop + edge counts) -> rsqrt in place
    gcn_deg_init<<<cdiv_ll(N, 256), 256, 0, stream>>>(inv, N);
    gcn_deg_count<<<cdiv_ll(E, 256), 256, 0, stream>>>(inv, dst, E);
    gcn_inv_sqrt<<<cdiv_ll(N, 256), 256, 0, stream>>>(inv, N);

    // 2. layer 1: t1 = x @ W1 -> P[:, :128]; h1 = Ahat @ t1 + b1 -> Q
    gcn_gemm_wmma<0, 0><<<dim3(cdiv_ll(N, 128), 128 / 64, 1), gblk, 0, stream>>>(
        x, W1p, nullptr, P, N, 128, 128);
    gcn_agg_init<<<cdiv_ll((long long)N * 32, 256), 256, 0, stream>>>(P, b1, inv, Q, N, 128);
    gcn_edge_scatter<<<cdiv_ll((long long)E * 32, 256), 256, 0, stream>>>(
        P, Q, src, dst, inv, E, 128);

    // 3. layer 2: t2 = relu(h1) @ W2 -> P[:, :256]; h2 -> Q
    gcn_gemm_wmma<1, 0><<<dim3(cdiv_ll(N, 128), 256 / 64, 1), gblk, 0, stream>>>(
        Q, W2p, nullptr, P, N, 128, 256);
    gcn_agg_init<<<cdiv_ll((long long)N * 64, 256), 256, 0, stream>>>(P, b2, inv, Q, N, 256);
    gcn_edge_scatter<<<cdiv_ll((long long)E * 64, 256), 256, 0, stream>>>(
        P, Q, src, dst, inv, E, 256);

    // 4. layer 3: t3 = relu(h2) @ W3 -> P[:, :512]; h3 -> Q (no relu)
    gcn_gemm_wmma<1, 0><<<dim3(cdiv_ll(N, 128), 512 / 64, 1), gblk, 0, stream>>>(
        Q, W3p, nullptr, P, N, 256, 512);
    gcn_agg_init<<<cdiv_ll((long long)N * 128, 256), 256, 0, stream>>>(P, b3, inv, Q, N, 512);
    gcn_edge_scatter<<<cdiv_ll((long long)E * 128, 256), 256, 0, stream>>>(
        P, Q, src, dst, inv, E, 512);

    // 5. MLP head: z1 = relu(h3@lw1+lb1) -> P[N,1024]; z2 = relu(z1@lw2+lb2) -> Q[N,512]
    gcn_gemm_wmma<0, 1><<<dim3(cdiv_ll(N, 128), 1024 / 64, 1), gblk, 0, stream>>>(
        Q, L1p, lb1, P, N, 512, 1024);
    gcn_gemm_wmma<0, 1><<<dim3(cdiv_ll(N, 128), 512 / 64, 1), gblk, 0, stream>>>(
        P, L2p, lb2, Q, N, 1024, 512);

    // 6. z3 = z2@lw3+lb3 fused with global mean pool
    gcn_zero<<<cdiv_ll(N_GRAPHS * 4 + N_GRAPHS, 256), 256, 0, stream>>>(
        sums, N_GRAPHS * 4 + N_GRAPHS);
    gcn_head_pool<<<cdiv_ll((long long)N * 32, 256), 256, 0, stream>>>(
        Q, lw3, lb3, bat, sums, counts, N);
    gcn_pool_div<<<cdiv_ll(N_GRAPHS * 4, 256), 256, 0, stream>>>(sums, counts, (float*)d_out);
}
